// VGGT_21122649161904
// MI455X (gfx1250) — matrix operand order repaired and checked by reading, not compiled, not run on hardware
//
#include <hip/hip_runtime.h>
#include <math.h>

typedef __attribute__((ext_vector_type(16))) _Float16 v16h;
typedef __attribute__((ext_vector_type(8)))  float    v8f;

#define CDIM   256
#define NTOK   4096            // N*S*P = 2*4*512
#define TTOT   2496            // sum of COUNTS
#define NEGV   (-1.0e9f)

__device__ __forceinline__ float gelu_tanh(float x) {
  return 0.5f * x * (1.0f + tanhf(0.7978845608028654f * (x + 0.044715f * x * x * x)));
}

// ---------------------------------------------------------------------------
// CDNA5 async global->LDS copy (ASYNCcnt-tracked), per-lane 4B scatter.
// lds_off is the workgroup-relative LDS byte address (flat addr low 32 bits).
// ---------------------------------------------------------------------------
__device__ __forceinline__ void async_b32(unsigned lds_off, const float* g) {
  asm volatile("global_load_async_to_lds_b32 %0, %1, off"
               :: "v"(lds_off), "v"((unsigned long long)(size_t)g)
               : "memory");
}
__device__ __forceinline__ void wait_async_le16() {
  asm volatile("s_wait_asynccnt 16" ::: "memory");
}
__device__ __forceinline__ void wait_async_le0() {
  asm volatile("s_wait_asynccnt 0" ::: "memory");
}

// ---------------------------------------------------------------------------
// Block-tiled WMMA GEMM: C[M,N] = act(A[M,K] @ B[K,N] + bias) (+ res)
// Workgroup = 128 threads (4 waves) computing a 64x64 tile; wave w owns the
// 16x64 strip at row m0+16w (4 accumulators, A fragment reused 4x).
// B chunk [32K x 64N] is staged in LDS (transposed to [n][K]) via
// global_load_async_to_lds_b32, double-buffered and pipelined with
// s_wait_asynccnt; fragments feed v_wmma_f32_16x16x32_f16 (f32 accumulate).
// ---------------------------------------------------------------------------
#define GEMM_BN 64

__global__ __launch_bounds__(128)
void gemm_wmma_async(const float* __restrict__ A, const float* __restrict__ B,
                     const float* __restrict__ bias, const float* __restrict__ res,
                     float* __restrict__ Cout, int M, int N, int K, int act) {
  const int tid  = threadIdx.x;
  const int lane = tid & 31;
  const int wid  = tid >> 5;
  const int n0   = blockIdx.x * GEMM_BN;
  const int m0   = blockIdx.y * 64 + wid * 16;
  const int hi   = lane >> 4;          // lane half: 0 = lanes 0-15, 1 = 16-31
  const int q    = lane & 15;

  __shared__ float sB[2][GEMM_BN * 32];   // double-buffered, layout [n][K]

  // Issue async fill of sB[buf] with B rows [k0, k0+32), cols [n0, n0+64).
  auto issue = [&](int buf, int k0) {
    const unsigned base = (unsigned)(size_t)(&sB[buf][0]);
    #pragma unroll
    for (int it = 0; it < 16; ++it) {
      int s  = it * 128 + tid;            // slot in [0, 2048)
      int nn = s >> 5;                    // column within tile
      int kk = s & 31;                    // K within chunk
      async_b32(base + (unsigned)(nn * 32 + kk) * 4u,
                B + (size_t)(k0 + kk) * N + (n0 + nn));
    }
  };

  v8f acc[4] = {};
  issue(0, 0);
  const int nsteps = K >> 5;
  for (int st = 0; st < nsteps; ++st) {
    const int cur = st & 1;
    if (st + 1 < nsteps) {
      issue(1 - cur, (st + 1) << 5);
      wait_async_le16();                  // in-order: previous 16 have landed
    } else {
      wait_async_le0();
    }
    __syncthreads();

    // A fragment (16x32): lane = row; K = {0-15} / {16-31} per half
    v16h afrag;
    const float* ap = A + (size_t)(m0 + q) * K + (st << 5) + (hi ? 16 : 0);
    #pragma unroll
    for (int r = 0; r < 8; ++r) {
      int kk = 2 * r;
      afrag[2 * r]     = (_Float16)ap[kk];
      afrag[2 * r + 1] = (_Float16)ap[kk + 1];
    }
    // 4 B fragments from LDS; transposed layout makes each lane's 16 K-values
    // a contiguous 64B run (vectorizable ds reads).
    const float* bb = &sB[cur][0] + (hi ? 16 : 0);
    #pragma unroll
    for (int j = 0; j < 4; ++j) {
      v16h bfrag;
      const float* bp = bb + (size_t)(j * 16 + q) * 32;
      #pragma unroll
      for (int e = 0; e < 16; ++e) bfrag[e] = (_Float16)bp[e];
      acc[j] = __builtin_amdgcn_wmma_f32_16x16x32_f16(false, afrag, false, bfrag,
                                                      (short)0, acc[j], false, false);
    }
    __syncthreads();                      // all waves done with sB[cur]
  }

  // Epilogue: C layout lane = col, VGPR i -> row i + 8*half
  const int rbase = hi ? 8 : 0;
  #pragma unroll
  for (int j = 0; j < 4; ++j) {
    const int   n  = n0 + j * 16 + q;
    const float bv = bias ? bias[n] : 0.0f;
    #pragma unroll
    for (int i = 0; i < 8; ++i) {
      int   m = m0 + rbase + i;
      float v = acc[j][i] + bv;
      if (act) v = gelu_tanh(v);
      if (res) v += res[(size_t)m * N + n];
      Cout[(size_t)m * N + n] = v;
    }
  }
}

// ---------------------------------------------------------------------------
// LayerNorm over last dim (256): one wave per row, shfl reductions (wave32).
// ---------------------------------------------------------------------------
__global__ __launch_bounds__(32)
void ln_kernel(const float* __restrict__ x, const float* __restrict__ g,
               const float* __restrict__ b, float* __restrict__ y) {
  const int row  = blockIdx.x;
  const int lane = threadIdx.x;
  const float* xp = x + (size_t)row * CDIM;
  float vals[8], s = 0.f, sq = 0.f;
  #pragma unroll
  for (int i = 0; i < 8; ++i) {
    vals[i] = xp[lane + 32 * i];
    s += vals[i];
    sq += vals[i] * vals[i];
  }
  #pragma unroll
  for (int off = 16; off >= 1; off >>= 1) {
    s  += __shfl_xor(s, off, 32);
    sq += __shfl_xor(sq, off, 32);
  }
  float mean = s * (1.0f / CDIM);
  float var  = sq * (1.0f / CDIM) - mean * mean;
  float inv  = rsqrtf(var + 1e-6f);
  #pragma unroll
  for (int i = 0; i < 8; ++i) {
    int c = lane + 32 * i;
    y[(size_t)row * CDIM + c] = (vals[i] - mean) * inv * g[c] + b[c];
  }
}

// ---------------------------------------------------------------------------
// Ragged unflatten + tokenizer add + positional embed (K=3, scalar math).
// ---------------------------------------------------------------------------
__global__ __launch_bounds__(256)
void embed_kernel(const int* __restrict__ counts, const float* __restrict__ coords,
                  const float* __restrict__ tokf, const float* __restrict__ posW,
                  const float* __restrict__ posb, float* __restrict__ x) {
  const int t  = blockIdx.x;
  const int c  = threadIdx.x;
  const int bf = t >> 9;         // (n*4 + s)
  const int p  = t & 511;
  int start = 0;
  #pragma unroll
  for (int i = 0; i < 8; ++i) start += (i < bf) ? counts[i] : 0;
  float v = 0.0f;
  if (p < counts[bf]) {
    int src = start + p;
    float pe = coords[src * 3 + 0] * posW[0 * CDIM + c]
             + coords[src * 3 + 1] * posW[1 * CDIM + c]
             + coords[src * 3 + 2] * posW[2 * CDIM + c] + posb[c];
    v = tokf[(size_t)src * CDIM + c] + pe;
  }
  x[(size_t)t * CDIM + c] = v;
}

// ---------------------------------------------------------------------------
// Fused flash-style attention: one wave per (batch, head, 16-query tile).
// DH=32 -> one WMMA per 16x16 score tile; online softmax; P@V via two WMMAs
// with accumulator rescale. Mask derived from patch_counts on the fly.
// ---------------------------------------------------------------------------
__global__ __launch_bounds__(32)
void attn_kernel(const float* __restrict__ qkv, const int* __restrict__ counts,
                 float* __restrict__ out, int L) {
  const int lane = threadIdx.x;
  const int q0   = blockIdx.x << 4;
  const int h    = blockIdx.y;
  const int tok0 = blockIdx.z * L;
  const int hi   = lane >> 4;
  const int qi   = lane & 15;
  const float scale = 0.17677669529663687f;  // 1/sqrt(32)

  __shared__ float sP[16 * 32];
  __shared__ float sM[16], sSum[16], sScale[16];
  if (lane < 16) { sM[lane] = -3.0e38f; sSum[lane] = 0.0f; }
  __syncthreads();

  // Q^T B-fragment (32 d x 16 queries), loaded once.
  v16h qfrag;
  {
    const float* qp = qkv + (size_t)(tok0 + q0 + qi) * 768 + h * 32 + (hi ? 16 : 0);
    #pragma unroll
    for (int r = 0; r < 8; ++r) {
      qfrag[2 * r]     = (_Float16)qp[2 * r];
      qfrag[2 * r + 1] = (_Float16)qp[2 * r + 1];
    }
  }

  v8f acc0 = {}, acc1 = {};
  const int rbase = hi ? 8 : 0;

  for (int kc = 0; kc < L; kc += 32) {
    // ---- two score WMMAs: S^T tiles (lane col = query) ----
    #pragma unroll
    for (int part = 0; part < 2; ++part) {
      v16h kfrag;  // A fragment: 16 keys x 32 d
      const float* kp = qkv + (size_t)(tok0 + kc + part * 16 + qi) * 768 + 256
                      + h * 32 + (hi ? 16 : 0);
      #pragma unroll
      for (int r = 0; r < 8; ++r) {
        int d = 2 * r;
        kfrag[2 * r]     = (_Float16)kp[d];
        kfrag[2 * r + 1] = (_Float16)kp[d + 1];
      }
      v8f c = {};
      c = __builtin_amdgcn_wmma_f32_16x16x32_f16(false, kfrag, false, qfrag,
                                                 (short)0, c, false, false);
      #pragma unroll
      for (int i = 0; i < 8; ++i) {
        int keyr = kc + part * 16 + rbase + i;
        int t    = tok0 + keyr;
        bool valid = (t & 511) < counts[t >> 9];
        sP[qi * 32 + part * 16 + rbase + i] = valid ? c[i] * scale : NEGV;
      }
    }
    __syncthreads();

    // ---- online softmax stats: lanes 0-15 each own one query row ----
    if (lane < 16) {
      float m_old = sM[lane], m_new = m_old;
      #pragma unroll
      for (int j = 0; j < 32; ++j) m_new = fmaxf(m_new, sP[lane * 32 + j]);
      float resc = __expf(m_old - m_new);
      float s = 0.0f;
      #pragma unroll
      for (int j = 0; j < 32; ++j) {
        float e = __expf(sP[lane * 32 + j] - m_new);
        sP[lane * 32 + j] = e;
        s += e;
      }
      sSum[lane]   = sSum[lane] * resc + s;
      sM[lane]     = m_new;
      sScale[lane] = resc;
    }
    __syncthreads();

    // ---- P @ V: A = P [16 q x 32 keys] from LDS; B = V chunk [32 x 32] ----
    v16h pfrag;
    {
      const float* pp = sP + qi * 32 + (hi ? 16 : 0);
      #pragma unroll
      for (int r = 0; r < 8; ++r) {
        int kk = 2 * r;
        pfrag[2 * r]     = (_Float16)pp[kk];
        pfrag[2 * r + 1] = (_Float16)pp[kk + 1];
      }
    }
    #pragma unroll
    for (int i = 0; i < 8; ++i) {
      float f = sScale[rbase + i];
      acc0[i] *= f;
      acc1[i] *= f;
    }
    #pragma unroll
    for (int dt = 0; dt < 2; ++dt) {
      v16h vfrag;  // B fragment: 32 keys x 16 d
      const float* vp = qkv + (size_t)(tok0 + kc + (hi ? 16 : 0)) * 768 + 512
                      + h * 32 + dt * 16 + qi;
      #pragma unroll
      for (int r = 0; r < 8; ++r) {
        vfrag[2 * r]     = (_Float16)vp[(size_t)(2 * r) * 768];
        vfrag[2 * r + 1] = (_Float16)vp[(size_t)(2 * r + 1) * 768];
      }
      if (dt == 0)
        acc0 = __builtin_amdgcn_wmma_f32_16x16x32_f16(false, pfrag, false, vfrag,
                                                      (short)0, acc0, false, false);
      else
        acc1 = __builtin_amdgcn_wmma_f32_16x16x32_f16(false, pfrag, false, vfrag,
                                                      (short)0, acc1, false, false);
    }
    __syncthreads();
  }

  #pragma unroll
  for (int i = 0; i < 8; ++i) {
    int   qrow = q0 + rbase + i;
    float inv  = 1.0f / sSum[rbase + i];
    float* op  = out + (size_t)(tok0 + qrow) * CDIM + h * 32;
    op[qi]      = acc0[i] * inv;
    op[16 + qi] = acc1[i] * inv;
  }
}

// ---------------------------------------------------------------------------
// Final padding mask: out = x * mask
// ---------------------------------------------------------------------------
__global__ __launch_bounds__(256)
void mask_kernel(const int* __restrict__ counts, const float* __restrict__ x,
                 float* __restrict__ out) {
  const int t = blockIdx.x;
  const int c = threadIdx.x;
  bool valid = (t & 511) < counts[t >> 9];
  out[(size_t)t * CDIM + c] = valid ? x[(size_t)t * CDIM + c] : 0.0f;
}

// ---------------------------------------------------------------------------
extern "C" void kernel_launch(void* const* d_in, const int* in_sizes, int n_in,
                              void* d_out, int out_size, void* d_ws, size_t ws_size,
                              hipStream_t stream) {
  (void)in_sizes; (void)n_in; (void)out_size; (void)ws_size;

  const int*   counts  = (const int*)d_in[0];
  const float* coords  = (const float*)d_in[1];
  const float* patches = (const float*)d_in[2];
  const float* tokW    = (const float*)d_in[3];
  const float* tokb    = (const float*)d_in[4];
  const float* posW    = (const float*)d_in[5];
  const float* posb    = (const float*)d_in[6];
  const float* fb[12];
  const float* gb[12];
  for (int i = 0; i < 12; ++i) fb[i] = (const float*)d_in[7 + i];
  for (int i = 0; i < 12; ++i) gb[i] = (const float*)d_in[19 + i];

  // workspace layout (f32)
  float* ws   = (float*)d_ws;
  float* tokf = ws;                               // [TTOT, 256]
  float* x    = tokf + (size_t)TTOT * CDIM;       // [4096, 256]
  float* hbuf = x    + (size_t)NTOK * CDIM;       // [4096, 256]
  float* qkv  = hbuf + (size_t)NTOK * CDIM;       // [4096, 768]
  float* attn = qkv  + (size_t)NTOK * 768;        // [4096, 256]
  float* hid  = attn + (size_t)NTOK * CDIM;       // [4096, 1024]

  // 1) tokenizer: [2496,1024] x [1024,256]
  gemm_wmma_async<<<dim3(CDIM / 64, TTOT / 64), 128, 0, stream>>>(
      patches, tokW, tokb, nullptr, tokf, TTOT, CDIM, 1024, 0);

  // 2) ragged unflatten + pos embed -> x [4096, 256]
  embed_kernel<<<NTOK, 256, 0, stream>>>(counts, coords, tokf, posW, posb, x);

  // 3) transformer blocks: frame (B=8, L=512), then global (B=2, L=2048)
  auto run_block = [&](const float* const* p, int L, int Bc) {
    ln_kernel<<<NTOK, 32, 0, stream>>>(x, p[0], p[1], hbuf);
    gemm_wmma_async<<<dim3(768 / 64, NTOK / 64), 128, 0, stream>>>(
        hbuf, p[2], p[3], nullptr, qkv, NTOK, 768, CDIM, 0);
    attn_kernel<<<dim3(L / 16, 8, Bc), 32, 0, stream>>>(qkv, counts, attn, L);
    gemm_wmma_async<<<dim3(CDIM / 64, NTOK / 64), 128, 0, stream>>>(
        attn, p[4], p[5], x, x, NTOK, CDIM, CDIM, 0);            // + residual
    ln_kernel<<<NTOK, 32, 0, stream>>>(x, p[6], p[7], hbuf);
    gemm_wmma_async<<<dim3(1024 / 64, NTOK / 64), 128, 0, stream>>>(
        hbuf, p[8], p[9], nullptr, hid, NTOK, 1024, CDIM, 1);    // gelu
    gemm_wmma_async<<<dim3(CDIM / 64, NTOK / 64), 128, 0, stream>>>(
        hid, p[10], p[11], x, x, NTOK, CDIM, 1024, 0);           // + residual
  };
  run_block(fb, 512, 8);
  run_block(gb, 2048, 2);

  // 4) final mask
  mask_kernel<<<NTOK, 256, 0, stream>>>(counts, x, (float*)d_out);
}
